// InverseResNet_927712936203
// MI455X (gfx1250) — matrix operand order repaired and checked by reading, not compiled, not run on hardware
//
#include <hip/hip_runtime.h>

#define LATENT  128
#define HIDDEN  256
#define OUTDIM  128
#define NBLOCKS 4
#define NITER   10
#define BM      32      // batch rows per workgroup
#define WSTR    272     // padded LDS k-stride for weights (bf16 elements)
#define XSTR    272     // padded LDS k-stride for activations

typedef __attribute__((ext_vector_type(16))) __bf16 v16bf;
typedef __attribute__((ext_vector_type(8)))  __bf16 v8bf;
typedef __attribute__((ext_vector_type(8)))  float  v8f;
typedef __attribute__((ext_vector_type(4)))  float  v4f;

// LDS layout (bytes): all offsets 16B aligned, total 315904 <= 320KB
#define W1T_OFF 0
#define W2T_OFF 139264
#define XS_OFF  278528
#define TS_OFF  295936
#define B1_OFF  313344
#define B2_OFF  314368
#define BF_OFF  315392
#define LDS_BYTES 315904

union AFrag { v16bf v; v8bf h[2]; };

// A fragment (16x32 bf16): lane L: M = rbase + (L&15);
// halves 0..7  -> K = k0 + 8*(L>>4) + [0..7]
// halves 8..15 -> K = k0 + 16 + 8*(L>>4) + [0..7]
__device__ __forceinline__ v16bf load_a(const __bf16* src, int rbase, int k0, int lane) {
  const __bf16* p = src + (rbase + (lane & 15)) * XSTR + k0 + ((lane >> 4) << 3);
  AFrag u;
  u.h[0] = *(const v8bf*)p;
  u.h[1] = *(const v8bf*)(p + 16);
  return u.v;
}

// B fragment (32x16 bf16): lane L: N = nbase + (L&15);
// halves 0..7  -> K = k0 + 8*(L>>4) + [0..7]; halves 8..15 -> K = k0 + 16 + 8*(L>>4) + [0..7]  (weights stored transposed: wt[n][k])
__device__ __forceinline__ v16bf load_b(const __bf16* wt, int nbase, int k0, int lane) {
  const __bf16* p = wt + (nbase + (lane & 15)) * WSTR + k0 + ((lane >> 4) << 3);
  AFrag u;
  u.h[0] = *(const v8bf*)p;
  u.h[1] = *(const v8bf*)(p + 16);
  return u.v;
}

// Cooperative load of row-major [K][N] f32 weights into LDS as transposed bf16 [N][WSTR]
__device__ __forceinline__ void load_wt(__bf16* dst, const float* __restrict__ src,
                                        int K, int N, int tid) {
  const int total = K * N;
  for (int base = tid * 4; base < total; base += 256 * 4) {
    int k = base / N;
    int n = base % N;                       // N multiple of 4 -> n multiple of 4
    v4f w = *(const v4f*)(src + base);      // coalesced along N
#pragma unroll
    for (int j = 0; j < 4; ++j)
      dst[(n + j) * WSTR + k] = (__bf16)w[j];
  }
}

// Store a 16x16 f32 accumulator tile to LDS as bf16 (C/D layout:
// VGPR v, lane L -> row rbase + 8*(L>>4) + v, col nbase + (L&15))
__device__ __forceinline__ void store_acc_bf16(__bf16* dst, int rbase, int nbase,
                                               v8f a, int lane) {
  int col = nbase + (lane & 15);
  int r0  = rbase + ((lane >> 4) << 3);
#pragma unroll
  for (int v = 0; v < 8; ++v)
    dst[(r0 + v) * XSTR + col] = (__bf16)a[v];
}

extern "C" __global__ __launch_bounds__(256)
void irn_fused(const float* __restrict__ x,     const float* __restrict__ Winit,
               const float* __restrict__ binit,
               const float* __restrict__ Wg1,   const float* __restrict__ bg1,
               const float* __restrict__ Wg2,   const float* __restrict__ bg2,
               const float* __restrict__ Wfin,  const float* __restrict__ bfin,
               float* __restrict__ out) {
  extern __shared__ char smem[];
  __bf16* w1t = (__bf16*)(smem + W1T_OFF);
  __bf16* w2t = (__bf16*)(smem + W2T_OFF);
  __bf16* xs  = (__bf16*)(smem + XS_OFF);
  __bf16* ts  = (__bf16*)(smem + TS_OFF);
  float*  b1s = (float*)(smem + B1_OFF);
  float*  b2s = (float*)(smem + B2_OFF);
  float*  bfs = (float*)(smem + BF_OFF);

  const int tid   = threadIdx.x;
  const int lane  = tid & 31;
  const int wave  = tid >> 5;
  const int cb    = wave << 5;          // wave's col base: 0,32,...,224
  const int lrow  = lane & 15;
  const int rhi   = (lane >> 4) << 3;   // accumulator row offset within 16-row tile
  const int rowg0 = blockIdx.x * BM;

  const v8f zf = {0.f, 0.f, 0.f, 0.f, 0.f, 0.f, 0.f, 0.f};

  // ---- stage 0: x tile (f32->bf16) into xs[:,0:128]; W_init^T into w1t ----
  {
    int row = tid >> 3;
    int c0  = (tid & 7) << 4;
    const float* src = x + (size_t)(rowg0 + row) * LATENT + c0;
#pragma unroll
    for (int j = 0; j < 4; ++j) {
      v4f w = *(const v4f*)(src + 4 * j);
#pragma unroll
      for (int e = 0; e < 4; ++e)
        xs[row * XSTR + c0 + 4 * j + e] = (__bf16)w[e];
    }
    load_wt(w1t, Winit, LATENT, HIDDEN, tid);
    if (tid < HIDDEN) b1s[tid] = binit[tid];
  }
  __syncthreads();

  // ---- stage 1: y = x @ W_init + b_init (K=128) ----
  // Each wave: 32x32 output tile -> acc[mi][ni], mi: rows 16*mi, ni: cols cb+16*ni
  v8f y[2][2];
#pragma unroll
  for (int mi = 0; mi < 2; ++mi)
#pragma unroll
    for (int ni = 0; ni < 2; ++ni) y[mi][ni] = zf;
#pragma unroll
  for (int k0 = 0; k0 < LATENT; k0 += 32) {
    v16bf a0 = load_a(xs, 0, k0, lane);
    v16bf a1 = load_a(xs, 16, k0, lane);
    v16bf b0 = load_b(w1t, cb, k0, lane);
    v16bf b1 = load_b(w1t, cb + 16, k0, lane);
    y[0][0] = __builtin_amdgcn_wmma_f32_16x16x32_bf16(false, a0, false, b0, (short)0, y[0][0], false, false);
    y[0][1] = __builtin_amdgcn_wmma_f32_16x16x32_bf16(false, a0, false, b1, (short)0, y[0][1], false, false);
    y[1][0] = __builtin_amdgcn_wmma_f32_16x16x32_bf16(false, a1, false, b0, (short)0, y[1][0], false, false);
    y[1][1] = __builtin_amdgcn_wmma_f32_16x16x32_bf16(false, a1, false, b1, (short)0, y[1][1], false, false);
  }
#pragma unroll
  for (int mi = 0; mi < 2; ++mi)
#pragma unroll
    for (int ni = 0; ni < 2; ++ni) y[mi][ni] += b1s[cb + 16 * ni + lrow];

  __syncthreads();                       // all reads of x-input done
#pragma unroll
  for (int mi = 0; mi < 2; ++mi)
#pragma unroll
    for (int ni = 0; ni < 2; ++ni)
      store_acc_bf16(xs, 16 * mi, cb + 16 * ni, y[mi][ni], lane);

  // ---- stage 2: 4 blocks x 10 fixed-point iterations ----
  v8f xc[2][2];
#pragma unroll
  for (int mi = 0; mi < 2; ++mi)
#pragma unroll
    for (int ni = 0; ni < 2; ++ni) xc[mi][ni] = y[mi][ni];

#pragma unroll 1
  for (int blk = 0; blk < NBLOCKS; ++blk) {
    __syncthreads();                     // previous readers of w1t/w2t done
    load_wt(w1t, Wg1 + (size_t)blk * HIDDEN * HIDDEN, HIDDEN, HIDDEN, tid);
    load_wt(w2t, Wg2 + (size_t)blk * HIDDEN * HIDDEN, HIDDEN, HIDDEN, tid);
    if (tid < HIDDEN) {
      b1s[tid] = bg1[blk * HIDDEN + tid];
      b2s[tid] = bg2[blk * HIDDEN + tid];
    }
    if (blk + 1 < NBLOCKS) {             // warm L2 for next block's weights
      __builtin_prefetch(Wg1 + (size_t)(blk + 1) * HIDDEN * HIDDEN + tid * 64, 0, 1);
      __builtin_prefetch(Wg2 + (size_t)(blk + 1) * HIDDEN * HIDDEN + tid * 64, 0, 1);
    }
    __syncthreads();

#pragma unroll 1
    for (int it = 0; it < NITER; ++it) {
      // t = relu(x @ W1 + b1)
      v8f acc[2][2];
#pragma unroll
      for (int mi = 0; mi < 2; ++mi)
#pragma unroll
        for (int ni = 0; ni < 2; ++ni) acc[mi][ni] = zf;
#pragma unroll
      for (int k0 = 0; k0 < HIDDEN; k0 += 32) {
        v16bf a0 = load_a(xs, 0, k0, lane);
        v16bf a1 = load_a(xs, 16, k0, lane);
        v16bf b0 = load_b(w1t, cb, k0, lane);
        v16bf b1 = load_b(w1t, cb + 16, k0, lane);
        acc[0][0] = __builtin_amdgcn_wmma_f32_16x16x32_bf16(false, a0, false, b0, (short)0, acc[0][0], false, false);
        acc[0][1] = __builtin_amdgcn_wmma_f32_16x16x32_bf16(false, a0, false, b1, (short)0, acc[0][1], false, false);
        acc[1][0] = __builtin_amdgcn_wmma_f32_16x16x32_bf16(false, a1, false, b0, (short)0, acc[1][0], false, false);
        acc[1][1] = __builtin_amdgcn_wmma_f32_16x16x32_bf16(false, a1, false, b1, (short)0, acc[1][1], false, false);
      }
#pragma unroll
      for (int mi = 0; mi < 2; ++mi) {
#pragma unroll
        for (int ni = 0; ni < 2; ++ni) {
          int col = cb + 16 * ni + lrow;
          float bias = b1s[col];
          int r0 = 16 * mi + rhi;
#pragma unroll
          for (int v = 0; v < 8; ++v) {
            float t = acc[mi][ni][v] + bias;
            t = t > 0.f ? t : 0.f;
            ts[(r0 + v) * XSTR + col] = (__bf16)t;
          }
        }
      }
      __syncthreads();

      // g = t @ W2 + b2 ; x <- y - g
#pragma unroll
      for (int mi = 0; mi < 2; ++mi)
#pragma unroll
        for (int ni = 0; ni < 2; ++ni) acc[mi][ni] = zf;
#pragma unroll
      for (int k0 = 0; k0 < HIDDEN; k0 += 32) {
        v16bf a0 = load_a(ts, 0, k0, lane);
        v16bf a1 = load_a(ts, 16, k0, lane);
        v16bf b0 = load_b(w2t, cb, k0, lane);
        v16bf b1 = load_b(w2t, cb + 16, k0, lane);
        acc[0][0] = __builtin_amdgcn_wmma_f32_16x16x32_bf16(false, a0, false, b0, (short)0, acc[0][0], false, false);
        acc[0][1] = __builtin_amdgcn_wmma_f32_16x16x32_bf16(false, a0, false, b1, (short)0, acc[0][1], false, false);
        acc[1][0] = __builtin_amdgcn_wmma_f32_16x16x32_bf16(false, a1, false, b0, (short)0, acc[1][0], false, false);
        acc[1][1] = __builtin_amdgcn_wmma_f32_16x16x32_bf16(false, a1, false, b1, (short)0, acc[1][1], false, false);
      }
#pragma unroll
      for (int mi = 0; mi < 2; ++mi) {
#pragma unroll
        for (int ni = 0; ni < 2; ++ni) {
          float bias = b2s[cb + 16 * ni + lrow];
#pragma unroll
          for (int v = 0; v < 8; ++v)
            xc[mi][ni][v] = y[mi][ni][v] - (acc[mi][ni][v] + bias);
          store_acc_bf16(xs, 16 * mi, cb + 16 * ni, xc[mi][ni], lane);
        }
      }
      __syncthreads();
    }
#pragma unroll
    for (int mi = 0; mi < 2; ++mi)
#pragma unroll
      for (int ni = 0; ni < 2; ++ni) y[mi][ni] = xc[mi][ni];   // next block's input
  }

  // ---- stage 3: out = h @ W_final + b_final (N=128, 16 cols per wave) ----
  load_wt(w1t, Wfin, HIDDEN, OUTDIM, tid);
  if (tid < OUTDIM) bfs[tid] = bfin[tid];
  __syncthreads();

  const int cb2 = wave << 4;   // 0,16,...,112
  v8f oacc[2];
  oacc[0] = zf;
  oacc[1] = zf;
#pragma unroll
  for (int k0 = 0; k0 < HIDDEN; k0 += 32) {
    v16bf a0 = load_a(xs, 0, k0, lane);
    v16bf a1 = load_a(xs, 16, k0, lane);
    v16bf b  = load_b(w1t, cb2, k0, lane);
    oacc[0] = __builtin_amdgcn_wmma_f32_16x16x32_bf16(false, a0, false, b, (short)0, oacc[0], false, false);
    oacc[1] = __builtin_amdgcn_wmma_f32_16x16x32_bf16(false, a1, false, b, (short)0, oacc[1], false, false);
  }
#pragma unroll
  for (int mi = 0; mi < 2; ++mi) {
    int col = cb2 + lrow;
    float bias = bfs[col];
    int r0 = rowg0 + 16 * mi + rhi;
#pragma unroll
    for (int v = 0; v < 8; ++v)
      out[(size_t)(r0 + v) * OUTDIM + col] = oacc[mi][v] + bias;
  }
}

extern "C" void kernel_launch(void* const* d_in, const int* in_sizes, int n_in,
                              void* d_out, int out_size, void* d_ws, size_t ws_size,
                              hipStream_t stream) {
  const float* x     = (const float*)d_in[0];
  const float* Winit = (const float*)d_in[1];
  const float* binit = (const float*)d_in[2];
  const float* Wg1   = (const float*)d_in[3];
  const float* bg1   = (const float*)d_in[4];
  const float* Wg2   = (const float*)d_in[5];
  const float* bg2   = (const float*)d_in[6];
  const float* Wfin  = (const float*)d_in[7];
  const float* bfin  = (const float*)d_in[8];
  float* out = (float*)d_out;

  const int batch = in_sizes[0] / LATENT;
  const int nblks = batch / BM;

  (void)hipFuncSetAttribute((const void*)irn_fused,
                            hipFuncAttributeMaxDynamicSharedMemorySize, LDS_BYTES);

  irn_fused<<<nblks, 256, LDS_BYTES, stream>>>(x, Winit, binit, Wg1, bg1, Wg2, bg2,
                                               Wfin, bfin, out);
}
